// fc_net_6253472382992
// MI455X (gfx1250) — matrix operand order repaired and checked by reading, not compiled, not run on hardware
//
#include <hip/hip_runtime.h>
#include <hip/hip_bf16.h>

#define NUM_GENO 16384
#define NUM_ENV  53
#define RDIM     2666
#define BATCH    256
#define EPSV     1e-5f

// ---- CDNA5 WMMA types (wave32) ----
typedef __attribute__((ext_vector_type(16))) __bf16        v16bf;
typedef __attribute__((ext_vector_type(8)))  float         v8f;
typedef __attribute__((ext_vector_type(4)))  unsigned int  u32x4;
typedef __attribute__((ext_vector_type(8)))  unsigned int  u32x8;

union Frag {
  v16bf v;
  u32x4 q[2];
  unsigned short u[16];
};

// hardware packed f32->bf16 conversion (round-to-nearest-even)
__device__ __forceinline__ unsigned cvt_pk_bf16(float lo, float hi) {
  unsigned r;
  asm("v_cvt_pk_bf16_f32 %0, %1, %2" : "=v"(r) : "v"(lo), "v"(hi));
  return r;
}
__device__ __forceinline__ unsigned short f2bf(float f) {
  return (unsigned short)(cvt_pk_bf16(f, f) & 0xffffu);
}
__device__ __forceinline__ float bf2f(unsigned short h) {
  union { unsigned int u; float f; } c; c.u = ((unsigned int)h) << 16;
  return c.f;
}

// ---------------------------------------------------------------------------
// Prep kernels: f32 -> bf16 activation staging
// ---------------------------------------------------------------------------
__global__ void convert_x_kernel(const float* __restrict__ x,
                                 unsigned short* __restrict__ xbf) {
  int e = blockIdx.x * blockDim.x + threadIdx.x;     // pair index
  if (e >= BATCH * NUM_GENO / 2) return;
  int g2 = e * 2;
  int b = g2 >> 14;                                  // NUM_GENO = 2^14
  int g = g2 & (NUM_GENO - 1);
  const float* p = x + (size_t)b * (NUM_GENO + NUM_ENV) + g;
  *(unsigned*)(xbf + g2) = cvt_pk_bf16(p[0], p[1]);  // xbf 4B-aligned (g even)
}

__global__ void prep_kernel(const float* __restrict__ x,
                            unsigned short* __restrict__ envbf,
                            unsigned short* __restrict__ flat) {
  int b = threadIdx.x;  // 256 threads, 1 block
  for (int f = 0; f < 64; ++f)
    envbf[b * 64 + f] = (f < NUM_ENV)
        ? f2bf(x[(size_t)b * (NUM_GENO + NUM_ENV) + NUM_GENO + f])
        : (unsigned short)0;
  for (int c = RDIM * 5; c < 13344; ++c)   // zero K-pad of flat buffer
    flat[(size_t)b * 13344 + c] = 0;
}

// ---------------------------------------------------------------------------
// Workhorse GEMM: C[M,N] = relu( bn( A[M,K](bf16) @ W[N,K](f32->bf16)^T + bias ) ) (+res)
// Block 128x128 (8 waves), wave tile 64x32 -> 8 x v_wmma_f32_16x16x32_bf16 / K-step.
// Double-buffered LDS pipeline: tile kt+1 staged (A via TDM async DMA; W via
// vectorized f32->bf16 stream) while tile kt is consumed by WMMA.
// ---------------------------------------------------------------------------
#define BM  128
#define BN  128
#define KS  32
#define LDT 40   // LDS row stride (bf16): 64B data + 16B TDM pad -> 80B, 16B aligned

__global__ __launch_bounds__(256)
void gemm_bf16_bn_relu(const unsigned short* __restrict__ A, int lda,
                       const float* __restrict__ W,
                       const float* __restrict__ bias,
                       const float* __restrict__ bng, const float* __restrict__ bnb,
                       const float* __restrict__ bnm, const float* __restrict__ bnv,
                       const unsigned short* __restrict__ res, int ldres,
                       unsigned short* __restrict__ outB, float* __restrict__ outF,
                       int ldc, int M, int N, int K) {
  __shared__ __align__(16) unsigned short AsmB[2][BM * LDT];
  __shared__ __align__(16) unsigned short WsmB[2][BN * LDT];

  const int t    = threadIdx.x;
  const int lane = t & 31;
  const int wv   = t >> 5;
  const int mw   = wv & 1;    // 2 M-waves -> *64
  const int nw   = wv >> 1;   // 4 N-waves -> *32
  const int m0   = blockIdx.y * BM;
  const int n0   = blockIdx.x * BN;
  const int l15  = lane & 15;
  const int kh   = lane >> 4;

  v8f acc[4][2] = {};
  const int kIters = (K + KS - 1) / KS;

  // ---- TDM issue: 2D async copy of 128 x 32 bf16 A tile, LDS-padded to LDT
  auto issueA = [&](int kt, int b) {
    if (wv != 0) return;
    const int k0 = kt * KS;
    const unsigned long long ga =
        (unsigned long long)(const void*)(A + (size_t)m0 * lda + k0);
    const unsigned ldsA = (unsigned)(unsigned long long)(const void*)AsmB[b];
    const unsigned td0  = (unsigned)(lda - k0);   // >= 32 (lda padded to 32)
    u32x4 g0;
    g0[0] = 1u;                                            // count=1, no gather
    g0[1] = ldsA;                                          // lds_addr
    g0[2] = (unsigned)(ga & 0xffffffffull);                // global_addr[31:0]
    g0[3] = (unsigned)((ga >> 32) & 0x1ffffffull) | (2u << 30); // addr[56:32]|type=2
    u32x8 g1;
    g1[0] = (1u << 16)            // data_size = 1 (2 bytes)
          | (1u << 20)            // pad_enable
          | (3u << 22)            // pad_interval: every 16 DWORDs (64B row)
          | (3u << 25);           // pad_amount: 4 DWORDs (16B) -> row stride 80B
    g1[1] = (td0 & 0xffffu) << 16;                         // tensor_dim0[15:0]
    g1[2] = (td0 >> 16) | (128u << 16);                    // td0[31:16] | td1=128
    g1[3] = (32u << 16);                                   // tile_dim0 = 32
    g1[4] = 128u;                                          // tile_dim1=128, tile_dim2=0
    g1[5] = (unsigned)lda;                                 // tensor_dim0_stride lo
    g1[6] = 0u;
    g1[7] = 0u;
    asm volatile("tensor_load_to_lds %0, %1" :: "s"(g0), "s"(g1) : "memory");
  };

  // ---- W tile staging: f32 stream -> bf16 LDS
  auto stageWt = [&](int kt, int b) {
    const int k0 = kt * KS;
    unsigned short* Ws = WsmB[b];
    if (k0 + KS < K)
      __builtin_prefetch(&W[(size_t)(n0 + (t >> 5)) * K + k0 + KS + (t & 31)], 0, 1);
    const bool fullTile = (n0 + BN <= N) && (k0 + KS <= K);
    if (fullTile) {
      // fast path: unguarded float2 loads (all K values even -> 8B aligned),
      // one v_cvt_pk_bf16_f32 + one 32-bit LDS store per pair
      const int n8 = t >> 4;          // 0..15
      const int kp = (t & 15) * 2;    // 0,2,..,30
#pragma unroll
      for (int p = 0; p < 8; ++p) {
        const int n = p * 16 + n8;
        const float2 w2 = *(const float2*)&W[(size_t)(n0 + n) * K + k0 + kp];
        *(unsigned*)&Ws[n * LDT + kp] = cvt_pk_bf16(w2.x, w2.y);
      }
    } else {
      // tail path: clamped (always-legal) loads + VALU select, no exec juggling
      const int wn = t >> 5;          // 0..7
      const int wk = t & 31;
      const int k  = k0 + wk;
      const int kc = k < K ? k : K - 1;
#pragma unroll
      for (int p = 0; p < 16; ++p) {
        const int n  = p * 8 + wn;
        const int nc = (n0 + n) < N ? (n0 + n) : N - 1;
        const float raw = W[(size_t)nc * K + kc];
        Ws[n * LDT + wk] = f2bf(((n0 + n) < N && k < K) ? raw : 0.0f);
      }
    }
  };

  // ---- pipeline prologue: stage tile 0 into buffer 0
  issueA(0, 0);
  stageWt(0, 0);
  if (wv == 0) __builtin_amdgcn_s_wait_tensorcnt(0);
  __syncthreads();

  for (int kt = 0; kt < kIters; ++kt) {
    const int cb = kt & 1;
    const int nb = cb ^ 1;
    const bool more = (kt + 1) < kIters;
    if (more) {
      issueA(kt + 1, nb);     // async DMA in flight during this tile's WMMAs
      stageWt(kt + 1, nb);
    }

    // ---- fragments in contiguous k order (k = 16h + s, same for A and B), then 8 WMMAs on current buffer
    const unsigned short* Ab = AsmB[cb];
    const unsigned short* Wb = WsmB[cb];
    Frag fa[4], fb[2];
#pragma unroll
    for (int fi = 0; fi < 4; ++fi) {
      const unsigned short* p = Ab + (mw * 64 + fi * 16 + l15) * LDT;
      fa[fi].q[0] = *(const u32x4*)(p + kh * 16);
      fa[fi].q[1] = *(const u32x4*)(p + kh * 16 + 8);
    }
#pragma unroll
    for (int fj = 0; fj < 2; ++fj) {
      const unsigned short* p = Wb + (nw * 32 + fj * 16 + l15) * LDT + kh * 16;
      fb[fj].q[0] = *(const u32x4*)(p);
      fb[fj].q[1] = *(const u32x4*)(p + 8);
    }
#pragma unroll
    for (int fi = 0; fi < 4; ++fi)
#pragma unroll
      for (int fj = 0; fj < 2; ++fj)
        acc[fi][fj] = __builtin_amdgcn_wmma_f32_16x16x32_bf16(
            false, fa[fi].v, false, fb[fj].v, (short)0, acc[fi][fj], false, false);

    if (more && wv == 0) __builtin_amdgcn_s_wait_tensorcnt(0);
    __syncthreads();   // publish buffer nb for next iteration
  }

  // ---- epilogue: bias + folded BN + relu + bf16 residual; bf16 or f32 store
#pragma unroll
  for (int fj = 0; fj < 2; ++fj) {
    const int col = n0 + nw * 32 + fj * 16 + l15;
    const bool colOK = col < N;
    const int colc = colOK ? col : N - 1;       // clamped, always-legal loads
    const float bi = colOK ? bias[colc] : 0.0f;
    float sc = 1.0f, mm = 0.0f, bb = 0.0f;
    const bool hasbn = (bng != nullptr);
    if (hasbn) {
      sc = colOK ? bng[colc] * rsqrtf(bnv[colc] + EPSV) : 1.0f;
      mm = colOK ? bnm[colc] : 0.0f;
      bb = colOK ? bnb[colc] : 0.0f;
    }
#pragma unroll
    for (int fi = 0; fi < 4; ++fi) {
      const int rowb = m0 + mw * 64 + fi * 16 + kh * 8;
#pragma unroll
      for (int r = 0; r < 8; ++r) {
        const int row = rowb + r;
        float v = acc[fi][fj][r] + bi;
        if (hasbn) v = (v - mm) * sc + bb;
        v = fmaxf(v, 0.0f);
        if (res) v += bf2f(res[(size_t)row * ldres + col]);
        if (outB)       outB[(size_t)row * ldc + col] = colOK ? f2bf(v) : (unsigned short)0;
        else if (colOK) outF[(size_t)row * ldc + col] = v;
      }
    }
  }
}

// ---------------------------------------------------------------------------
// Fused per-locus head: for each i -> z=[env|h[:,i]], then
//   d3 = z@Wlin[i]^T+blin[i];  h3 = relu(bn3_j( . @W13/W14/W15^T + b ))
// Four 256x64x64 WMMA GEMMs entirely in LDS; writes only (256,5) to flat.
// ---------------------------------------------------------------------------
__global__ __launch_bounds__(256)
void head_fused_kernel(const unsigned short* __restrict__ envbf,
                       const unsigned short* __restrict__ hbf,  // 256 x 2688 bf16
                       const float* __restrict__ Wlin, const float* __restrict__ blin,
                       const float* __restrict__ W13,  const float* __restrict__ b13,
                       const float* __restrict__ W14,  const float* __restrict__ b14,
                       const float* __restrict__ W15,  const float* __restrict__ b15,
                       const float* __restrict__ bn3g, const float* __restrict__ bn3b,
                       const float* __restrict__ bn3m, const float* __restrict__ bn3v,
                       unsigned short* __restrict__ flat) {     // 256 x 13344 bf16
  __shared__ __align__(16) unsigned short Ash[BATCH * 64];  // activations, in-place
  __shared__ __align__(16) unsigned short Wsh[64 * 64];     // current weight tile

  const int i    = blockIdx.x;
  const int t    = threadIdx.x;
  const int lane = t & 31;
  const int wv   = t >> 5;       // wave owns rows [wv*32, wv*32+32)
  const int l15  = lane & 15;
  const int kh   = lane >> 4;

  // build z = [env(53) | h[:,i] | 0-pad] into LDS
  {
    const u32x4* src = (const u32x4*)(envbf + (size_t)t * 64);
    u32x4* dst = (u32x4*)(Ash + t * 64);
#pragma unroll
    for (int q = 0; q < 8; ++q) dst[q] = src[q];
    Ash[t * 64 + NUM_ENV] = hbf[(size_t)t * 2688 + i];
  }

  auto stageW = [&](const float* Wg, int nr, int kc) {
    const int n  = t >> 2;
    const int kb = (t & 3) * 16;
#pragma unroll
    for (int j = 0; j < 16; ++j) {
      const int k = kb + j;
      const int nc = n < nr ? n : nr - 1;
      const int kcl = k < kc ? k : kc - 1;
      const float raw = Wg[nc * kc + kcl];
      Wsh[n * 64 + k] = f2bf((n < nr && k < kc) ? raw : 0.0f);
    }
  };

  auto doGemm = [&](v8f acc[2][4]) {
#pragma unroll
    for (int ks = 0; ks < 2; ++ks) {
      const int k0 = ks * 32;
      Frag fa[2], fb[4];
#pragma unroll
      for (int fi = 0; fi < 2; ++fi) {
        const unsigned short* p = Ash + (wv * 32 + fi * 16 + l15) * 64 + k0;
        fa[fi].q[0] = *(const u32x4*)(p + kh * 16);
        fa[fi].q[1] = *(const u32x4*)(p + kh * 16 + 8);
      }
#pragma unroll
      for (int fj = 0; fj < 4; ++fj) {
        const unsigned short* p = Wsh + (fj * 16 + l15) * 64 + k0 + kh * 16;
        fb[fj].q[0] = *(const u32x4*)(p);
        fb[fj].q[1] = *(const u32x4*)(p + 8);
      }
#pragma unroll
      for (int fi = 0; fi < 2; ++fi)
#pragma unroll
        for (int fj = 0; fj < 4; ++fj)
          acc[fi][fj] = __builtin_amdgcn_wmma_f32_16x16x32_bf16(
              false, fa[fi].v, false, fb[fj].v, (short)0, acc[fi][fj], false, false);
    }
  };

  // ---- stage 0: d3 = z @ Wlin[i]^T + blin[i]   (no bn/relu)
  stageW(Wlin + (size_t)i * 54 * 54, 54, 54);
  __syncthreads();
  {
    v8f acc[2][4] = {};
    doGemm(acc);
#pragma unroll
    for (int fj = 0; fj < 4; ++fj) {
      const int col = fj * 16 + l15;
      const float bi = (col < 54) ? blin[(size_t)i * 54 + col] : 0.0f;
#pragma unroll
      for (int fi = 0; fi < 2; ++fi) {
        const int rowb = wv * 32 + fi * 16 + kh * 8;
#pragma unroll
        for (int r = 0; r < 8; ++r)
          Ash[(rowb + r) * 64 + col] = f2bf(acc[fi][fj][r] + bi);  // wave-private rows
      }
    }
  }
  __syncthreads();

  // ---- stages 1..2: W13 (bn3[0]) and W14 (bn3[1])
  const float* Wmat[2] = { W13, W14 };
  const float* bvec[2] = { b13, b14 };
#pragma unroll 1
  for (int s = 0; s < 2; ++s) {
    stageW(Wmat[s], 54, 54);
    __syncthreads();
    const float sc = bn3g[s * RDIM + i] * rsqrtf(bn3v[s * RDIM + i] + EPSV);
    const float mm = bn3m[s * RDIM + i];
    const float be = bn3b[s * RDIM + i];
    v8f acc[2][4] = {};
    doGemm(acc);
#pragma unroll
    for (int fj = 0; fj < 4; ++fj) {
      const int col = fj * 16 + l15;
      const float bi = (col < 54) ? bvec[s][col] : 0.0f;
#pragma unroll
      for (int fi = 0; fi < 2; ++fi) {
        const int rowb = wv * 32 + fi * 16 + kh * 8;
#pragma unroll
        for (int r = 0; r < 8; ++r) {
          float v = ((acc[fi][fj][r] + bi) - mm) * sc + be;
          v = fmaxf(v, 0.0f);
          Ash[(rowb + r) * 64 + col] = (col < 54) ? f2bf(v) : (unsigned short)0;
        }
      }
    }
    __syncthreads();
  }

  // ---- stage 3: W15 (5x54), bn3[2], write final (256,5) slice to flat
  stageW(W15, 5, 54);
  __syncthreads();
  {
    const float sc = bn3g[2 * RDIM + i] * rsqrtf(bn3v[2 * RDIM + i] + EPSV);
    const float mm = bn3m[2 * RDIM + i];
    const float be = bn3b[2 * RDIM + i];
    v8f acc[2][4] = {};
    doGemm(acc);
    const int col = l15;               // only fj==0, cols 0..4 matter
    if (col < 5) {
      const float bi = b15[col];
#pragma unroll
      for (int fi = 0; fi < 2; ++fi) {
        const int rowb = wv * 32 + fi * 16 + kh * 8;
#pragma unroll
        for (int r = 0; r < 8; ++r) {
          float v = ((acc[fi][0][r] + bi) - mm) * sc + be;
          v = fmaxf(v, 0.0f);
          flat[(size_t)(rowb + r) * 13344 + i * 5 + col] = f2bf(v);
        }
      }
    }
  }
}

// ---------------------------------------------------------------------------
// Host orchestration
// ---------------------------------------------------------------------------
extern "C" void kernel_launch(void* const* d_in, const int* in_sizes, int n_in,
                              void* d_out, int out_size, void* d_ws, size_t ws_size,
                              hipStream_t stream) {
  const float* x    = (const float*)d_in[0];
  const float* W1   = (const float*)d_in[1];
  const float* b1   = (const float*)d_in[2];
  const float* Wmid = (const float*)d_in[3];
  const float* bmid = (const float*)d_in[4];
  const float* bng  = (const float*)d_in[5];
  const float* bnb  = (const float*)d_in[6];
  const float* bnm  = (const float*)d_in[7];
  const float* bnv  = (const float*)d_in[8];
  const float* Wlin = (const float*)d_in[9];
  const float* blin = (const float*)d_in[10];
  const float* W13  = (const float*)d_in[11];
  const float* b13  = (const float*)d_in[12];
  const float* W14  = (const float*)d_in[13];
  const float* b14  = (const float*)d_in[14];
  const float* W15  = (const float*)d_in[15];
  const float* b15  = (const float*)d_in[16];
  const float* bn3g = (const float*)d_in[17];
  const float* bn3b = (const float*)d_in[18];
  const float* bn3m = (const float*)d_in[19];
  const float* bn3v = (const float*)d_in[20];
  const float* W16  = (const float*)d_in[21];
  const float* b16  = (const float*)d_in[22];
  const float* W17  = (const float*)d_in[23];
  const float* b17  = (const float*)d_in[24];
  const float* W18  = (const float*)d_in[25];
  const float* b18  = (const float*)d_in[26];
  float* out = (float*)d_out;
  (void)in_sizes; (void)n_in; (void)out_size; (void)ws_size;

  // workspace layout (all bf16 activations, K-strides padded to 32, ~21 MB)
  char* ws = (char*)d_ws;
  size_t off = 0;
  auto grab = [&](size_t bytes) -> char* {
    char* p = ws + off;
    off += (bytes + 255) & ~(size_t)255;
    return p;
  };
  unsigned short* xbf  = (unsigned short*)grab((size_t)BATCH * NUM_GENO * 2);
  unsigned short* envb = (unsigned short*)grab((size_t)BATCH * 64 * 2);
  unsigned short* h[3];
  for (int s = 0; s < 3; ++s) h[s] = (unsigned short*)grab((size_t)BATCH * 2688 * 2);
  unsigned short* flat = (unsigned short*)grab((size_t)BATCH * 13344 * 2);
  unsigned short* g1   = (unsigned short*)grab((size_t)BATCH * 768 * 2);
  unsigned short* g2   = (unsigned short*)grab((size_t)BATCH * 768 * 2);

  convert_x_kernel<<<(BATCH * NUM_GENO / 2) / 256, 256, 0, stream>>>(x, xbf);
  prep_kernel<<<1, 256, 0, stream>>>(x, envb, flat);

  const dim3 gridR(2688 / 128, BATCH / 128);   // N=2666 -> 21 N-blocks
  // layer 1: h = relu(bn0(x_geno @ W1^T + b1))
  gemm_bf16_bn_relu<<<gridR, 256, 0, stream>>>(
      xbf, NUM_GENO, W1, b1, bng, bnb, bnm, bnv,
      nullptr, 0, h[0], nullptr, 2688, BATCH, RDIM, NUM_GENO);

  // 11 mid layers with residual every 2nd layer (buffers rotate over 3 slots)
  int cur = 0, dres = 0;
  for (int k = 0; k < 11; ++k) {
    const int next = (cur == dres) ? (cur + 1) % 3 : (3 - cur - dres);
    const unsigned short* rptr = (k % 2 == 1 && k < 10) ? h[dres] : nullptr;
    gemm_bf16_bn_relu<<<gridR, 256, 0, stream>>>(
        h[cur], 2688,
        Wmid + (size_t)k * RDIM * RDIM, bmid + (size_t)k * RDIM,
        bng + (size_t)(k + 1) * RDIM, bnb + (size_t)(k + 1) * RDIM,
        bnm + (size_t)(k + 1) * RDIM, bnv + (size_t)(k + 1) * RDIM,
        rptr, 2688, h[next], nullptr, 2688, BATCH, RDIM, RDIM);
    if (rptr) dres = next;
    cur = next;
  }

  // fused per-locus head (2666 workgroups, 4 LDS-resident WMMA GEMMs each)
  head_fused_kernel<<<RDIM, 256, 0, stream>>>(
      envb, h[cur], Wlin, blin, W13, b13, W14, b14, W15, b15,
      bn3g, bn3b, bn3m, bn3v, flat);

  // tail MLP: 13330 -> 750 -> 750 -> 750
  const dim3 grid750(768 / 128, BATCH / 128);
  gemm_bf16_bn_relu<<<grid750, 256, 0, stream>>>(
      flat, 13344, W16, b16, nullptr, nullptr, nullptr, nullptr,
      nullptr, 0, g1, nullptr, 768, BATCH, 750, RDIM * 5);
  gemm_bf16_bn_relu<<<grid750, 256, 0, stream>>>(
      g1, 768, W17, b17, nullptr, nullptr, nullptr, nullptr,
      nullptr, 0, g2, nullptr, 768, BATCH, 750, 750);
  gemm_bf16_bn_relu<<<grid750, 256, 0, stream>>>(
      g2, 768, W18, b18, nullptr, nullptr, nullptr, nullptr,
      nullptr, 0, nullptr, out, 750, BATCH, 750, 750);
}
